// InformationalAttention_38354057953723
// MI455X (gfx1250) — compile-verified, matrix operand order checked by reading, not run on hardware
//
#include <hip/hip_runtime.h>
#include <math.h>

#define DEVINL __device__ __forceinline__

typedef __attribute__((ext_vector_type(16))) __bf16          bf16x16;
typedef __attribute__((ext_vector_type(8)))  float           f32x8;
typedef __attribute__((ext_vector_type(4)))  float           f32x4;
typedef __attribute__((ext_vector_type(8)))  unsigned short  u16x8;
typedef __attribute__((ext_vector_type(16))) unsigned short  u16x16;
typedef __attribute__((ext_vector_type(4)))  unsigned int    u32x4;
typedef __attribute__((ext_vector_type(8)))  int             i32x8;
typedef __attribute__((ext_vector_type(4)))  int             i32x4;

constexpr int Tn  = 2048;
constexpr int Cn  = 2048;
constexpr int Hn  = 16;
constexpr int Dn  = 128;
constexpr int Rn  = 64;
constexpr int TP  = Tn + 32;     // key-padded length: 32-key tiles never read OOB
constexpr int WLOC = 512;

// ---------------------------------------------------------------------------
// helpers
// ---------------------------------------------------------------------------
DEVINL unsigned short f2bf(float f) {
  union { float f; unsigned u; } v; v.f = f;
  unsigned u = v.u;
  return (unsigned short)((u + 0x7FFFu + ((u >> 16) & 1u)) >> 16); // RNE
}

// A/B fragment for V_WMMA_F32_16X16X32_BF16 (ISA 7.12.2 16-bit layout):
// lane l owns row/col (l&15); lanes 0-15 hold K {0..7,16..23}, lanes 16-31
// hold K {8..15,24..31}.
DEVINL bf16x16 load_frag_u16(const unsigned short* row, int lane) {
  const int khalf = (lane >> 4) << 3;
  u16x8 lo = *(const u16x8*)(row + khalf);
  u16x8 hi = *(const u16x8*)(row + khalf + 16);
  u16x16 v;
#pragma unroll
  for (int i = 0; i < 8; ++i) { v[i] = lo[i]; v[8 + i] = hi[i]; }
  return __builtin_bit_cast(bf16x16, v);
}

DEVINL f32x8 wmma_bf16(bf16x16 a, bf16x16 b, f32x8 c) {
  return __builtin_amdgcn_wmma_f32_16x16x32_bf16(
      false, a, false, b, (short)0, c, false, false);
}

// ---------------------------------------------------------------------------
// Tensor Data Mover: 2D bf16 tile (rows x 32) global -> LDS with padding so
// the LDS row stride becomes 40 u16 (64B data + 16B pad), matching the
// fragment loader.  D# layout per CDNA5 ISA ch.8 (group0/group1, 2D tensor).
// ---------------------------------------------------------------------------
DEVINL void tdm_load_tile(unsigned lds_off, const unsigned short* gptr,
                          int rows, long lda_elems) {
  unsigned long long ga = (unsigned long long)(size_t)gptr;
  u32x4 g0;
  g0[0] = 1u;                                    // count=1, user mode
  g0[1] = lds_off;                               // LDS byte address
  g0[2] = (unsigned)(ga & 0xffffffffu);          // global_addr[31:0]
  g0[3] = (unsigned)((ga >> 32) & 0x1ffffffu)    // global_addr[56:32]
          | (2u << 30);                          // type = 2 ("image")
  unsigned long long st0 = (unsigned long long)lda_elems;
  const unsigned td0 = 32, tile0 = 32;
  const unsigned td1 = (unsigned)rows, tile1 = (unsigned)rows;
  i32x8 g1;
  g1[0] = (int)((1u << 16)        // data_size = 2 bytes
              | (1u << 20)        // pad_enable
              | (3u << 22)        // pad_interval: 16 DWORDs (64B)
              | (3u << 25));      // pad_amount:   4 DWORDs (16B)
  g1[1] = (int)((td0 & 0xffffu) << 16);                    // tensor_dim0[15:0]
  g1[2] = (int)((td0 >> 16) | ((td1 & 0xffffu) << 16));    // dim0 hi | dim1 lo
  g1[3] = (int)((td1 >> 16) | (tile0 << 16));              // dim1 hi | tile0
  g1[4] = (int)tile1;                                      // tile1 | tile2=0
  g1[5] = (int)(st0 & 0xffffffffu);                        // dim0_stride lo
  g1[6] = (int)((st0 >> 32) & 0xffffu);                    // dim0_stride hi
  g1[7] = 0;
  i32x4 z;
  z[0] = 0; z[1] = 0; z[2] = 0; z[3] = 0;
  i32x8 z8;
#pragma unroll
  for (int i = 0; i < 8; ++i) z8[i] = 0;
  __builtin_amdgcn_tensor_load_to_lds(g0, g1, z, z, z8, 0);
}

// ---------------------------------------------------------------------------
// bf16 tiled GEMM:  C[M,N] = op( A[M,K](bf16) * B[N,K](bf16)^T )
// Block tile 64x128, 8 waves (2x4), each wave a 32x32 tile = 4 WMMAs/K-step.
// Double-buffered TDM staging: wave 0 issues the next tile pair, then
// s_wait_tensorcnt 2 retires only the current pair (TDM ops complete in
// order within a wave), overlapping DMA with WMMA compute.
// mode: 0 f32 store, 1 f32 GELU, 2 f32 C += gate[m]*acc,
//       4 bf16 store, 5 bf16 transposed store
// ---------------------------------------------------------------------------
__global__ __launch_bounds__(256) void gemm_tdm_kernel(
    const unsigned short* __restrict__ A, int lda, long sA,
    const unsigned short* __restrict__ B, int ldb, long sB,
    void* __restrict__ Cv, int ldc, long sC,
    int N, int K, int mode, const float* __restrict__ gate) {
  __shared__ __align__(16) unsigned short As[2][64 * 40];
  __shared__ __align__(16) unsigned short Bs[2][128 * 40];

  A += (size_t)blockIdx.z * sA;
  B += (size_t)blockIdx.z * sB;

  const int m0 = blockIdx.y * 64;
  const int n0 = blockIdx.x * 128;
  const int nrows = (N - n0 < 128) ? (N - n0) : 128;
  const int tid = threadIdx.x;
  const int lane = tid & 31;
  const int wave = tid >> 5;
  const int wm = (wave >> 2) * 32;   // 0 / 32
  const int wn = (wave & 3) * 32;    // 0 / 32 / 64 / 96
  const int mrow = lane & 15;

  const unsigned ldsA0 = (unsigned)(size_t)(void*)As[0];  // flat->LDS addr[31:0]
  const unsigned ldsA1 = (unsigned)(size_t)(void*)As[1];
  const unsigned ldsB0 = (unsigned)(size_t)(void*)Bs[0];
  const unsigned ldsB1 = (unsigned)(size_t)(void*)Bs[1];

  f32x8 acc[2][2];
#pragma unroll
  for (int i = 0; i < 2; ++i)
#pragma unroll
    for (int j = 0; j < 2; ++j)
#pragma unroll
      for (int r = 0; r < 8; ++r) acc[i][j][r] = 0.f;

  if (wave == 0) {                      // prologue: stage K-tile 0 into buf 0
    tdm_load_tile(ldsA0, A + (size_t)m0 * lda, 64, lda);
    tdm_load_tile(ldsB0, B + (size_t)n0 * ldb, nrows, ldb);
  }

  for (int k0 = 0; k0 < K; k0 += 32) {
    const int cur = (k0 >> 5) & 1;
    if (wave == 0) {
      if (k0 + 32 < K) {                // stage next K-tile into other buffer
        tdm_load_tile(cur ? ldsA0 : ldsA1,
                      A + (size_t)m0 * lda + k0 + 32, 64, lda);
        tdm_load_tile(cur ? ldsB0 : ldsB1,
                      B + (size_t)n0 * ldb + k0 + 32, nrows, ldb);
        __builtin_amdgcn_s_wait_tensorcnt(2);  // current pair retired
      } else {
        __builtin_amdgcn_s_wait_tensorcnt(0);
      }
    }
    __syncthreads();
    const unsigned short* Ac = As[cur];
    const unsigned short* Bc = Bs[cur];
    bf16x16 a0 = load_frag_u16(Ac + (wm + mrow) * 40, lane);
    bf16x16 a1 = load_frag_u16(Ac + (wm + 16 + mrow) * 40, lane);
    bf16x16 b0 = load_frag_u16(Bc + (wn + mrow) * 40, lane);
    bf16x16 b1 = load_frag_u16(Bc + (wn + 16 + mrow) * 40, lane);
    acc[0][0] = wmma_bf16(a0, b0, acc[0][0]);
    acc[0][1] = wmma_bf16(a0, b1, acc[0][1]);
    acc[1][0] = wmma_bf16(a1, b0, acc[1][0]);
    acc[1][1] = wmma_bf16(a1, b1, acc[1][1]);
    __syncthreads();
  }

  float* Cf = (float*)Cv + (size_t)blockIdx.z * sC;
  unsigned short* Cu = (unsigned short*)Cv + (size_t)blockIdx.z * sC;
  const int mb = (lane >> 4) << 3;
#pragma unroll
  for (int i = 0; i < 2; ++i) {
#pragma unroll
    for (int j = 0; j < 2; ++j) {
      const int n = n0 + wn + j * 16 + mrow;
      if (n >= N) continue;
#pragma unroll
      for (int r = 0; r < 8; ++r) {
        const int m = m0 + wm + i * 16 + mb + r;
        float v = acc[i][j][r];
        if (mode == 1) {
          v = 0.5f * v * (1.f + erff(v * 0.70710678118654752f));
          Cf[(size_t)m * ldc + n] = v;
        } else if (mode == 2) {
          Cf[(size_t)m * ldc + n] += gate[m] * v;
        } else if (mode == 4) {
          Cu[(size_t)m * ldc + n] = f2bf(v);
        } else if (mode == 5) {
          Cu[(size_t)n * ldc + m] = f2bf(v);
        } else {
          Cf[(size_t)m * ldc + n] = v;
        }
      }
    }
  }
}

// ---------------------------------------------------------------------------
// fp32 -> bf16 flat convert
// ---------------------------------------------------------------------------
__global__ void cvt_f2bf_kernel(const float* __restrict__ src,
                                unsigned short* __restrict__ dst, int n) {
  int i = blockIdx.x * 256 + threadIdx.x;
  if (i < n) dst[i] = f2bf(src[i]);
}

// ---------------------------------------------------------------------------
// RoPE in-place on q and k slices of qkv[T, 3C] (fp32)
// ---------------------------------------------------------------------------
__global__ void rope_kernel(float* __restrict__ qkv) {
  int idx = blockIdx.x * blockDim.x + threadIdx.x;     // 2*T*H*64 = 2^22
  int i    = idx & 63;
  int h    = (idx >> 6) & (Hn - 1);
  int t    = (idx >> 10) & (Tn - 1);
  int part = idx >> 21;             // 0 = q, 1 = k
  float* p = qkv + (size_t)t * (3 * Cn) + part * Cn + h * Dn + 2 * i;
  float x0 = p[0], x1 = p[1];
  float f0 = __powf(10000.f, -(float)((2 * i) & 63) * (1.f / 64.f));
  float f1 = __powf(10000.f, -(float)((2 * i + 1) & 63) * (1.f / 64.f));
  float s0, c0, s1, c1;
  __sincosf((float)t * f0, &s0, &c0);
  __sincosf((float)t * f1, &s1, &c1);
  p[0] = x0 * c0 - x1 * s0;
  p[1] = x1 * c1 + x0 * s1;
}

// ---------------------------------------------------------------------------
// Pack roped qkv into bf16 attention layouts:
//   qbf[h][t][d], kbf[h][t(pad TP)][d], vbf[h][t][d], vT[h][d][t(pad TP)]
// ---------------------------------------------------------------------------
__global__ void pack_kernel(const float* __restrict__ qkv,
                            unsigned short* __restrict__ qbf,
                            unsigned short* __restrict__ kbf,
                            unsigned short* __restrict__ vbf,
                            unsigned short* __restrict__ vT) {
  int idx = blockIdx.x * blockDim.x + threadIdx.x;     // H*T*D = 2^22
  int d = idx & (Dn - 1);
  int t = (idx >> 7) & (Tn - 1);
  int h = idx >> 18;
  const float* row = qkv + (size_t)t * (3 * Cn) + h * Dn + d;
  qbf[(size_t)h * Tn * Dn + (size_t)t * Dn + d] = f2bf(row[0]);
  kbf[(size_t)h * TP * Dn + (size_t)t * Dn + d] = f2bf(row[Cn]);
  vbf[(size_t)h * Tn * Dn + (size_t)t * Dn + d] = f2bf(row[2 * Cn]);
  vT [(size_t)h * Dn * TP + (size_t)d * TP + t] = f2bf(row[2 * Cn]);
}

// ---------------------------------------------------------------------------
// gate[t] = sigmoid(h_info[t,:] . w2) > 0.75
// ---------------------------------------------------------------------------
__global__ __launch_bounds__(256) void info_gate_kernel(
    const float* __restrict__ h_info, const float* __restrict__ w2,
    float* __restrict__ gate) {
  __shared__ float red[256];
  const int t = blockIdx.x;
  float s = 0.f;
  for (int j = threadIdx.x; j < Cn / 4; j += 256)
    s += h_info[(size_t)t * (Cn / 4) + j] * w2[j];
  red[threadIdx.x] = s;
  __syncthreads();
  for (int st = 128; st > 0; st >>= 1) {
    if (threadIdx.x < st) red[threadIdx.x] += red[threadIdx.x + st];
    __syncthreads();
  }
  if (threadIdx.x == 0) {
    float sc = 1.f / (1.f + __expf(-red[0]));
    gate[t] = (sc > 0.75f) ? 1.f : 0.f;
  }
}

// ---------------------------------------------------------------------------
// Local windowed flash attention (window 512).  One wave per 16-query tile,
// 32-key steps, D=128.  Writes ctx[T, C] fp32.
// ---------------------------------------------------------------------------
__global__ __launch_bounds__(256) void local_attn_kernel(
    const unsigned short* __restrict__ qbf,
    const unsigned short* __restrict__ kbf,
    const unsigned short* __restrict__ vT,
    float* __restrict__ ctx) {
  __shared__ __align__(16) unsigned short Ps[8][16 * 40];
  const int h = blockIdx.y;
  const int wave = threadIdx.x >> 5;
  const int lane = threadIdx.x & 31;
  const int qt = (blockIdx.x * 8 + wave) * 16;
  const unsigned short* qh = qbf + (size_t)h * Tn * Dn;
  const unsigned short* kh = kbf + (size_t)h * TP * Dn;
  const unsigned short* vh = vT  + (size_t)h * Dn * TP;
  const int mrow = lane & 15;
  const int mb = (lane >> 4) << 3;

  bf16x16 qa[4];
#pragma unroll
  for (int s = 0; s < 4; ++s)
    qa[s] = load_frag_u16(qh + (size_t)(qt + mrow) * Dn + s * 32, lane);

  f32x8 o[8];
  float mi[8], li[8];
#pragma unroll
  for (int j = 0; j < 8; ++j)
#pragma unroll
    for (int r = 0; r < 8; ++r) o[j][r] = 0.f;
#pragma unroll
  for (int r = 0; r < 8; ++r) { mi[r] = -1e30f; li[r] = 0.f; }

  int s_begin = qt - (WLOC - 1); if (s_begin < 0) s_begin = 0;
  const float scale = 0.088388347648318447f;   // 1/sqrt(128)

  for (int kt = s_begin & ~31; kt < qt + 16; kt += 32) {
    f32x8 s0, s1;
#pragma unroll
    for (int r = 0; r < 8; ++r) { s0[r] = 0.f; s1[r] = 0.f; }
#pragma unroll
    for (int s = 0; s < 4; ++s) {
      bf16x16 b0 = load_frag_u16(kh + (size_t)(kt + mrow) * Dn + s * 32, lane);
      bf16x16 b1 = load_frag_u16(kh + (size_t)(kt + 16 + mrow) * Dn + s * 32, lane);
      s0 = wmma_bf16(qa[s], b0, s0);
      s1 = wmma_bf16(qa[s], b1, s1);
    }
    const int k0i = kt + mrow, k1i = k0i + 16;
    float p0[8], p1[8], rm[8];
#pragma unroll
    for (int r = 0; r < 8; ++r) {
      int tq = qt + mb + r;
      p0[r] = (k0i <= tq && k0i >= tq - (WLOC - 1)) ? s0[r] * scale : -1e30f;
      p1[r] = (k1i <= tq && k1i >= tq - (WLOC - 1)) ? s1[r] * scale : -1e30f;
      rm[r] = fmaxf(p0[r], p1[r]);
    }
#pragma unroll
    for (int r = 0; r < 8; ++r)
#pragma unroll
      for (int off = 1; off < 16; off <<= 1)
        rm[r] = fmaxf(rm[r], __shfl_xor(rm[r], off, 32));
    float corr[8], rs[8];
#pragma unroll
    for (int r = 0; r < 8; ++r) {
      float nm = fmaxf(mi[r], rm[r]);
      corr[r] = __expf(mi[r] - nm);
      mi[r] = nm;
      p0[r] = __expf(p0[r] - nm);
      p1[r] = __expf(p1[r] - nm);
      rs[r] = p0[r] + p1[r];
    }
#pragma unroll
    for (int r = 0; r < 8; ++r) {
#pragma unroll
      for (int off = 1; off < 16; off <<= 1)
        rs[r] += __shfl_xor(rs[r], off, 32);
      li[r] = li[r] * corr[r] + rs[r];
    }
#pragma unroll
    for (int j = 0; j < 8; ++j)
#pragma unroll
      for (int r = 0; r < 8; ++r) o[j][r] *= corr[r];

    unsigned short* ps = Ps[wave];
#pragma unroll
    for (int r = 0; r < 8; ++r) {
      int m = mb + r;
      ps[m * 40 + mrow]      = f2bf(p0[r]);
      ps[m * 40 + 16 + mrow] = f2bf(p1[r]);
    }
    asm volatile("" ::: "memory");   // in-wave DS ordering; block IR reorder
    bf16x16 pa = load_frag_u16(ps + mrow * 40, lane);
#pragma unroll
    for (int j = 0; j < 8; ++j) {
      bf16x16 vb = load_frag_u16(vh + (size_t)(j * 16 + mrow) * TP + kt, lane);
      o[j] = wmma_bf16(pa, vb, o[j]);
    }
  }
#pragma unroll
  for (int j = 0; j < 8; ++j)
#pragma unroll
    for (int r = 0; r < 8; ++r) {
      int tq = qt + mb + r;
      ctx[(size_t)tq * Cn + h * Dn + j * 16 + mrow] = o[j][r] / li[r];
    }
}

// ---------------------------------------------------------------------------
// Global low-rank flash attention over full T (no mask), rank R=64.
// All-bf16: q_r/k_r [H][T][R], v_rT [H][R][T] in, ctx_g_r [H][T][R] out.
// ---------------------------------------------------------------------------
__global__ __launch_bounds__(256) void global_attn_kernel(
    const unsigned short* __restrict__ q_r,
    const unsigned short* __restrict__ k_r,
    const unsigned short* __restrict__ v_rT,
    unsigned short* __restrict__ ctx_g_r) {
  __shared__ __align__(16) unsigned short Ps[8][16 * 40];
  const int h = blockIdx.y;
  const int wave = threadIdx.x >> 5;
  const int lane = threadIdx.x & 31;
  const int qt = (blockIdx.x * 8 + wave) * 16;
  const unsigned short* qh = q_r + (size_t)h * Tn * Rn;
  const unsigned short* kh = k_r + (size_t)h * Tn * Rn;
  const unsigned short* vh = v_rT + (size_t)h * Rn * Tn;
  const int mrow = lane & 15;
  const int mb = (lane >> 4) << 3;

  bf16x16 qa[2];
#pragma unroll
  for (int s = 0; s < 2; ++s)
    qa[s] = load_frag_u16(qh + (size_t)(qt + mrow) * Rn + s * 32, lane);

  f32x8 o[4];
  float mi[8], li[8];
#pragma unroll
  for (int j = 0; j < 4; ++j)
#pragma unroll
    for (int r = 0; r < 8; ++r) o[j][r] = 0.f;
#pragma unroll
  for (int r = 0; r < 8; ++r) { mi[r] = -1e30f; li[r] = 0.f; }

  const float scale = 0.17677669529663689f;    // (1/sqrt(D)) * D / R

  for (int kt = 0; kt < Tn; kt += 32) {
    f32x8 s0, s1;
#pragma unroll
    for (int r = 0; r < 8; ++r) { s0[r] = 0.f; s1[r] = 0.f; }
#pragma unroll
    for (int s = 0; s < 2; ++s) {
      bf16x16 b0 = load_frag_u16(kh + (size_t)(kt + mrow) * Rn + s * 32, lane);
      bf16x16 b1 = load_frag_u16(kh + (size_t)(kt + 16 + mrow) * Rn + s * 32, lane);
      s0 = wmma_bf16(qa[s], b0, s0);
      s1 = wmma_bf16(qa[s], b1, s1);
    }
    float p0[8], p1[8], rm[8];
#pragma unroll
    for (int r = 0; r < 8; ++r) {
      p0[r] = s0[r] * scale;
      p1[r] = s1[r] * scale;
      rm[r] = fmaxf(p0[r], p1[r]);
    }
#pragma unroll
    for (int r = 0; r < 8; ++r)
#pragma unroll
      for (int off = 1; off < 16; off <<= 1)
        rm[r] = fmaxf(rm[r], __shfl_xor(rm[r], off, 32));
    float corr[8], rs[8];
#pragma unroll
    for (int r = 0; r < 8; ++r) {
      float nm = fmaxf(mi[r], rm[r]);
      corr[r] = __expf(mi[r] - nm);
      mi[r] = nm;
      p0[r] = __expf(p0[r] - nm);
      p1[r] = __expf(p1[r] - nm);
      rs[r] = p0[r] + p1[r];
    }
#pragma unroll
    for (int r = 0; r < 8; ++r) {
#pragma unroll
      for (int off = 1; off < 16; off <<= 1)
        rs[r] += __shfl_xor(rs[r], off, 32);
      li[r] = li[r] * corr[r] + rs[r];
    }
#pragma unroll
    for (int j = 0; j < 4; ++j)
#pragma unroll
      for (int r = 0; r < 8; ++r) o[j][r] *= corr[r];

    unsigned short* ps = Ps[wave];
#pragma unroll
    for (int r = 0; r < 8; ++r) {
      int m = mb + r;
      ps[m * 40 + mrow]      = f2bf(p0[r]);
      ps[m * 40 + 16 + mrow] = f2bf(p1[r]);
    }
    asm volatile("" ::: "memory");
    bf16x16 pa = load_frag_u16(ps + mrow * 40, lane);
#pragma unroll
    for (int j = 0; j < 4; ++j) {
      bf16x16 vb = load_frag_u16(vh + (size_t)(j * 16 + mrow) * Tn + kt, lane);
      o[j] = wmma_bf16(pa, vb, o[j]);
    }
  }
#pragma unroll
  for (int j = 0; j < 4; ++j)
#pragma unroll
    for (int r = 0; r < 8; ++r) {
      int tq = qt + mb + r;
      ctx_g_r[(size_t)h * Tn * Rn + (size_t)tq * Rn + j * 16 + mrow] =
          f2bf(o[j][r] / li[r]);
    }
}

// ---------------------------------------------------------------------------
// Host launcher
// ---------------------------------------------------------------------------
extern "C" void kernel_launch(void* const* d_in, const int* in_sizes, int n_in,
                              void* d_out, int out_size, void* d_ws, size_t ws_size,
                              hipStream_t stream) {
  const float* x       = (const float*)d_in[0];
  const float* w_qkv   = (const float*)d_in[1];
  const float* w_o     = (const float*)d_in[2];
  const float* pk      = (const float*)d_in[3];
  const float* pv      = (const float*)d_in[4];
  const float* uo      = (const float*)d_in[5];
  const float* info_w1 = (const float*)d_in[6];
  const float* info_w2 = (const float*)d_in[7];
  float* out = (float*)d_out;

  char* ws = (char*)d_ws;
  size_t off = 0;
  auto carve = [&](size_t bytes) {
    void* p = ws + off;
    off = (off + bytes + 255) & ~(size_t)255;
    return p;
  };
  float*          qkv     = (float*)carve((size_t)Tn * 3 * Cn * 4);
  float*          h_info  = (float*)carve((size_t)Tn * (Cn / 4) * 4);
  float*          gate    = (float*)carve((size_t)Tn * 4);
  float*          ctx     = (float*)carve((size_t)Tn * Cn * 4);
  unsigned short* xbf     = (unsigned short*)carve((size_t)Tn * Cn * 2);
  unsigned short* wqkvbf  = (unsigned short*)carve((size_t)3 * Cn * Cn * 2);
  unsigned short* wobf    = (unsigned short*)carve((size_t)Cn * Cn * 2);
  unsigned short* iw1bf   = (unsigned short*)carve((size_t)(Cn / 4) * Cn * 2);
  unsigned short* pkbf    = (unsigned short*)carve((size_t)Rn * Dn * 2);
  unsigned short* pvbf    = (unsigned short*)carve((size_t)Rn * Dn * 2);
  unsigned short* uobf    = (unsigned short*)carve((size_t)Dn * Rn * 2);
  unsigned short* ctxbf   = (unsigned short*)carve((size_t)Tn * Cn * 2);
  unsigned short* qbf     = (unsigned short*)carve((size_t)Hn * Tn * Dn * 2);
  unsigned short* kbf     = (unsigned short*)carve((size_t)Hn * TP * Dn * 2);
  unsigned short* vbf     = (unsigned short*)carve((size_t)Hn * Tn * Dn * 2);
  unsigned short* vT      = (unsigned short*)carve((size_t)Hn * Dn * TP * 2);
  unsigned short* q_r     = (unsigned short*)carve((size_t)Hn * Tn * Rn * 2);
  unsigned short* k_r     = (unsigned short*)carve((size_t)Hn * Tn * Rn * 2);
  unsigned short* v_rT    = (unsigned short*)carve((size_t)Hn * Rn * Tn * 2);
  unsigned short* ctx_g_r = (unsigned short*)carve((size_t)Hn * Tn * Rn * 2);

  dim3 blk(256);
  auto cvt = [&](const float* s, unsigned short* d, int n) {
    cvt_f2bf_kernel<<<(n + 255) / 256, blk, 0, stream>>>(s, d, n);
  };

  // 0) one-time bf16 packs of all GEMM operands
  cvt(x, xbf, Tn * Cn);
  cvt(w_qkv, wqkvbf, 3 * Cn * Cn);
  cvt(w_o, wobf, Cn * Cn);
  cvt(info_w1, iw1bf, (Cn / 4) * Cn);
  cvt(pk, pkbf, Rn * Dn);
  cvt(pv, pvbf, Rn * Dn);
  cvt(uo, uobf, Dn * Rn);

  // 1) qkv = x @ w_qkv^T                 (2048 x 6144 x 2048) -> f32
  gemm_tdm_kernel<<<dim3(3 * Cn / 128, Tn / 64, 1), blk, 0, stream>>>(
      xbf, Cn, 0, wqkvbf, Cn, 0, qkv, 3 * Cn, 0, 3 * Cn, Cn, 0, nullptr);

  // 2) RoPE in place on q,k (f32)
  rope_kernel<<<(2 * Tn * Hn * 64) / 256, blk, 0, stream>>>(qkv);

  // 3) pack roped qkv -> bf16 attention layouts
  pack_kernel<<<(Hn * Tn * Dn) / 256, blk, 0, stream>>>(qkv, qbf, kbf, vbf, vT);

  // 4) h_info = gelu(x @ info_w1^T)      (2048 x 512 x 2048) -> f32
  gemm_tdm_kernel<<<dim3((Cn / 4) / 128, Tn / 64, 1), blk, 0, stream>>>(
      xbf, Cn, 0, iw1bf, Cn, 0, h_info, Cn / 4, 0, Cn / 4, Cn, 1, nullptr);

  // 5) gate[t]
  info_gate_kernel<<<Tn, blk, 0, stream>>>(h_info, info_w2, gate);

  // 6) local windowed attention -> ctx (f32)
  local_attn_kernel<<<dim3(Tn / 128, Hn), blk, 0, stream>>>(qbf, kbf, vT, ctx);

  // 7) q_r[h] = q[h] @ pk^T  (2048 x 64 x 128, batched heads) -> bf16
  gemm_tdm_kernel<<<dim3(1, Tn / 64, Hn), blk, 0, stream>>>(
      qbf, Dn, (long)Tn * Dn, pkbf, Dn, 0, q_r, Rn, (long)Tn * Rn,
      Rn, Dn, 4, nullptr);
  // 8) k_r[h] = k[h] @ pk^T -> bf16
  gemm_tdm_kernel<<<dim3(1, Tn / 64, Hn), blk, 0, stream>>>(
      kbf, Dn, (long)TP * Dn, pkbf, Dn, 0, k_r, Rn, (long)Tn * Rn,
      Rn, Dn, 4, nullptr);
  // 9) v_rT[h] = (v[h] @ pv^T)^T -> bf16 transposed store
  gemm_tdm_kernel<<<dim3(1, Tn / 64, Hn), blk, 0, stream>>>(
      vbf, Dn, (long)Tn * Dn, pvbf, Dn, 0, v_rT, Tn, (long)Rn * Tn,
      Rn, Dn, 5, nullptr);

  // 10) global low-rank attention -> ctx_g_r (bf16)
  global_attn_kernel<<<dim3(Tn / 128, Hn), blk, 0, stream>>>(q_r, k_r, v_rT,
                                                             ctx_g_r);

  // 11) ctx[:, h*D:] += gate[t] * (ctx_g_r[h] @ uo^T)   (2048 x 128 x 64)
  gemm_tdm_kernel<<<dim3(1, Tn / 64, Hn), blk, 0, stream>>>(
      ctx_g_r, Rn, (long)Tn * Rn, uobf, Rn, 0, ctx, Cn, (long)Dn,
      Dn, Rn, 2, gate);

  // 12) pack ctx -> bf16, then out = ctx @ w_o^T  (2048 x 2048 x 2048)
  cvt(ctx, ctxbf, Tn * Cn);
  gemm_tdm_kernel<<<dim3(Cn / 128, Tn / 64, 1), blk, 0, stream>>>(
      ctxbf, Cn, 0, wobf, Cn, 0, out, Cn, 0, Cn, Cn, 0, nullptr);

  (void)in_sizes; (void)n_in; (void)out_size; (void)ws_size;
}
